// entropy_72791105732799
// MI455X (gfx1250) — matrix operand order repaired and checked by reading, not compiled, not run on hardware
//
#include <hip/hip_runtime.h>

typedef unsigned short u16;
typedef __attribute__((ext_vector_type(16))) __bf16 v16bf;
typedef __attribute__((ext_vector_type(8)))  float  v8f;

#define N_ROWS 8192
#define N_COLS 8192
#define DIM    768
#define LDA    776            // padded LDS row stride in halfs (16B pad -> conflict-free b128 reads)
#define RPB    128            // rows staged per block (128*776*2 = 198656 B LDS)
#define NSTRIP 8              // column strips (grid parallelism)
#define CPS    (N_COLS / NSTRIP)   // 1024 cols per strip
#define NPART  (NSTRIP * 4)        // partial row-max slabs (strip x col-phase)

__device__ __forceinline__ u16 f32_to_bf16(float f) {
  union { float f; unsigned u; } v; v.f = f;
  unsigned r = v.u + 0x7FFFu + ((v.u >> 16) & 1u);   // round-to-nearest-even
  return (u16)(r >> 16);
}

union Frag { uint4 q[2]; v16bf v; };

__device__ __forceinline__ v8f v8f_zero() {
  v8f z;
#pragma unroll
  for (int i = 0; i < 8; ++i) z[i] = 0.0f;
  return z;
}

// ---------------- Kernel 1: fp32 -> bf16 + norms ----------------
__global__ __launch_bounds__(256) void prep_kernel(
    const float* __restrict__ ex, const float* __restrict__ ey,
    u16* __restrict__ exb, u16* __restrict__ eyb,
    float* __restrict__ nx, float* __restrict__ iny)
{
  __shared__ float red[256];
  const int row = blockIdx.x;
  const bool isx = row < N_ROWS;
  const int  r   = isx ? row : row - N_ROWS;
  const float* src = (isx ? ex : ey) + (size_t)r * DIM;
  u16*         dst = (isx ? exb : eyb) + (size_t)r * DIM;

  float ss = 0.0f;
  for (int c = threadIdx.x; c < DIM; c += 256) {
    float v = src[c];
    dst[c] = f32_to_bf16(v);
    ss += v * v;
  }
  red[threadIdx.x] = ss;
  __syncthreads();
  for (int s = 128; s > 0; s >>= 1) {
    if (threadIdx.x < s) red[threadIdx.x] += red[threadIdx.x + s];
    __syncthreads();
  }
  if (threadIdx.x == 0) {
    float nrm = sqrtf(red[0]);
    if (isx) nx[r]  = nrm;
    else     iny[r] = 1.0f / fmaxf(nrm, 1e-8f);
  }
}

// ---------------- Kernel 2: bf16 WMMA GEMM fused with row-max ----------------
__global__ __launch_bounds__(256) void gemm_rowmax_kernel(
    const u16* __restrict__ exb, const u16* __restrict__ eyb,
    const float* __restrict__ iny, float* __restrict__ rmax_part)
{
  extern __shared__ u16 lds_a[];   // [RPB][LDA]

  const int tid   = threadIdx.x;
  const int blk   = blockIdx.x;
  const int strip = blk & (NSTRIP - 1);
  const int rblk  = blk / NSTRIP;
  const int r0    = rblk * RPB;

  // Async (ASYNCcnt-tracked) stage of the 128x768 bf16 A stripe into LDS.
  // Each lane DMAs 16B directly from global to its LDS slot — no VGPR round trip.
  for (int idx = tid; idx < RPB * (DIM / 8); idx += 256) {
    int r  = idx / (DIM / 8);
    int c8 = idx - r * (DIM / 8);
    const u16* g = exb + (size_t)(r0 + r) * DIM + c8 * 8;
    unsigned ldsoff = (unsigned)(uintptr_t)(lds_a + r * LDA + c8 * 8);
    asm volatile("global_load_async_to_lds_b128 %0, %1, off"
                 :: "v"(ldsoff), "v"(g) : "memory");
  }
  asm volatile("s_wait_asynccnt 0x0" ::: "memory");
  __syncthreads();

  const int wave = tid >> 5;
  const int lane = tid & 31;
  const int rw   = wave & 1;       // row half: 64 rows each
  const int cw   = wave >> 1;      // column-tile phase 0..3
  const int l15  = lane & 15;
  const int half = lane >> 4;      // 0: lanes 0-15, 1: lanes 16-31
  const int hi8  = half * 8;       // A-frag K sub-offset / C row offset
  const int kb16 = half * 8;       // B-frag K offset

  const u16* abase = lds_a + (rw * 64) * LDA + l15 * LDA + hi8;

  float rmax[4][8];
#pragma unroll
  for (int rt = 0; rt < 4; ++rt)
#pragma unroll
    for (int v = 0; v < 8; ++v) rmax[rt][v] = -3.4e38f;

#pragma unroll 1
  for (int ct = cw; ct < CPS / 32; ct += 4) {
    const int c0 = strip * CPS + ct * 32;
    const float iny0 = iny[c0 + l15];
    const float iny1 = iny[c0 + 16 + l15];

    v8f acc[4][2];
#pragma unroll
    for (int rt = 0; rt < 4; ++rt) { acc[rt][0] = v8f_zero(); acc[rt][1] = v8f_zero(); }

    const u16* bcol0 = eyb + (size_t)(c0 + l15) * DIM + kb16;
    const u16* bcol1 = bcol0 + (size_t)16 * DIM;

#pragma unroll 1
    for (int k0 = 0; k0 < DIM; k0 += 32) {
      // Launder the K offset so the A-fragment LDS loads cannot be hoisted
      // out of the column-tile loop (that hoist caused scratch spills).
      int kk = k0;
      asm volatile("" : "+v"(kk));

      Frag b0, b1;
      b0.q[0] = *(const uint4*)(bcol0 + k0);
      b0.q[1] = *(const uint4*)(bcol0 + k0 + 16);
      b1.q[0] = *(const uint4*)(bcol1 + k0);
      b1.q[1] = *(const uint4*)(bcol1 + k0 + 16);
#pragma unroll
      for (int rt = 0; rt < 4; ++rt) {
        Frag a;
        const u16* ap = abase + rt * (16 * LDA) + kk;
        a.q[0] = *(const uint4*)ap;
        a.q[1] = *(const uint4*)(ap + 16);      // K +16..+23 half
        acc[rt][0] = __builtin_amdgcn_wmma_f32_16x16x32_bf16(
            false, a.v, false, b0.v, (short)0, acc[rt][0], false, false);
        acc[rt][1] = __builtin_amdgcn_wmma_f32_16x16x32_bf16(
            false, a.v, false, b1.v, (short)0, acc[rt][1], false, false);
      }
    }

#pragma unroll
    for (int rt = 0; rt < 4; ++rt)
#pragma unroll
      for (int v = 0; v < 8; ++v) {
        rmax[rt][v] = fmaxf(rmax[rt][v], acc[rt][0][v] * iny0);
        rmax[rt][v] = fmaxf(rmax[rt][v], acc[rt][1][v] * iny1);
      }
  }

  // Cross-lane max within each 16-lane half (rows 0-7 in lanes 0-15, 8-15 in 16-31).
#pragma unroll
  for (int rt = 0; rt < 4; ++rt)
#pragma unroll
    for (int v = 0; v < 8; ++v) {
      float m = rmax[rt][v];
#pragma unroll
      for (int off = 8; off > 0; off >>= 1)
        m = fmaxf(m, __shfl_xor(m, off, 32));
      if (l15 == 0) {
        int row = r0 + rw * 64 + rt * 16 + hi8 + v;
        rmax_part[(size_t)(strip * 4 + cw) * N_ROWS + row] = m;
      }
    }
}

// ---------------- Kernel 3: combine partials, log-prob, scalar sum ----------------
__global__ __launch_bounds__(256) void finalize_kernel(
    const float* __restrict__ rmax_part, const float* __restrict__ nx,
    float* __restrict__ out)
{
  __shared__ float red[256];
  float s = 0.0f;
  for (int row = threadIdx.x; row < N_ROWS; row += 256) {
    float m = -3.4e38f;
#pragma unroll
    for (int p = 0; p < NPART; ++p)
      m = fmaxf(m, rmax_part[(size_t)p * N_ROWS + row]);
    float c  = m / fmaxf(nx[row], 1e-8f);
    float z  = (c - 1.0f) * (1.0f / 0.3f);
    float lp = -0.5f * z * z + 0.28503426145f;   // -log(0.3) - 0.5*log(2*pi)
    s += expf(lp) * lp;
  }
  red[threadIdx.x] = s;
  __syncthreads();
  for (int t = 128; t > 0; t >>= 1) {
    if (threadIdx.x < t) red[threadIdx.x] += red[threadIdx.x + t];
    __syncthreads();
  }
  if (threadIdx.x == 0) out[0] = red[0];
}

// ---------------- Host launch ----------------
extern "C" void kernel_launch(void* const* d_in, const int* in_sizes, int n_in,
                              void* d_out, int out_size, void* d_ws, size_t ws_size,
                              hipStream_t stream) {
  const float* ex = (const float*)d_in[0];
  const float* ey = (const float*)d_in[1];

  char* ws = (char*)d_ws;
  size_t off = 0;
  u16* exb = (u16*)(ws + off);   off += (size_t)N_ROWS * DIM * sizeof(u16);
  u16* eyb = (u16*)(ws + off);   off += (size_t)N_COLS * DIM * sizeof(u16);
  float* nx  = (float*)(ws + off); off += (size_t)N_ROWS * sizeof(float);
  float* iny = (float*)(ws + off); off += (size_t)N_COLS * sizeof(float);
  float* rmax_part = (float*)(ws + off); off += (size_t)NPART * N_ROWS * sizeof(float);

  prep_kernel<<<N_ROWS + N_COLS, 256, 0, stream>>>(ex, ey, exb, eyb, nx, iny);

  const size_t lds_bytes = (size_t)RPB * LDA * sizeof(u16);   // 198656 B (< 320KB WGP LDS)
  gemm_rowmax_kernel<<<(N_ROWS / RPB) * NSTRIP, 256, lds_bytes, stream>>>(
      exb, eyb, iny, rmax_part);

  finalize_kernel<<<1, 256, 0, stream>>>(rmax_part, nx, (float*)d_out);
}
